// SpatialAttentionModule_25228637897510
// MI455X (gfx1250) — compile-verified, matrix operand order checked by reading, not run on hardware
//
#include <hip/hip_runtime.h>
#include <hip/hip_bf16.h>

#define B_   4
#define N_   2048
#define C_   256
#define H_   8
#define HD_  32
#define KE_  64      // extended head dim: 32 qk dims + 6 graph-bias dims + padding
#define BN_  6

typedef __attribute__((ext_vector_type(16))) __bf16 v16bf;
typedef __attribute__((ext_vector_type(8)))  float  v8f;

// Load a WMMA 16-bit A/B operand half-pair: 8 bf16 at p0, 8 bf16 at p1 (each 16B aligned).
__device__ __forceinline__ v16bf ld_pair(const __hip_bfloat16* p0, const __hip_bfloat16* p1) {
    union { v16bf v; uint4 q[2]; } u;
    u.q[0] = *(const uint4*)p0;
    u.q[1] = *(const uint4*)p1;
    return u.v;
}

__device__ __forceinline__ v8f wmma_bf16(v16bf a, v16bf b, v8f c) {
    return __builtin_amdgcn_wmma_f32_16x16x32_bf16(false, a, false, b, (short)0, c, false, false);
}

__device__ __forceinline__ __bf16 to_bf16(float f) {
    __hip_bfloat16 h = __float2bfloat16(f);
    __bf16 b;
    __builtin_memcpy(&b, &h, 2);
    return b;
}

// ---------------------------------------------------------------------------
// Kernel A: LayerNorm + bottleneck QKV (256->18->768), pack Q/K into extended
// bf16 layout [b][h][n][64] with graph-bias factors folded into dims 32..37,
// and V transposed to [b][h][dim][n]. One wave per token row.
// ---------------------------------------------------------------------------
__global__ __launch_bounds__(128) void qkv_pack_kernel(
    const float* __restrict__ x, const float* __restrict__ gamma, const float* __restrict__ beta,
    const float* __restrict__ wlow, const float* __restrict__ blow,
    const float* __restrict__ whigh, const float* __restrict__ bhigh,
    const float* __restrict__ u, const float* __restrict__ v,
    __hip_bfloat16* __restrict__ Qe, __hip_bfloat16* __restrict__ Ke,
    __hip_bfloat16* __restrict__ Vt)
{
    const int lane = threadIdx.x & 31;
    const int wid  = blockIdx.x * (blockDim.x >> 5) + (threadIdx.x >> 5);
    const int b = wid / N_;
    const int n = wid % N_;

    const float* xr = x + (size_t)(b * N_ + n) * C_;
    float xv[8], s = 0.f, s2 = 0.f;
#pragma unroll
    for (int i = 0; i < 8; i++) {
        float t = xr[lane + i * 32];
        xv[i] = t; s += t; s2 += t * t;
    }
#pragma unroll
    for (int off = 16; off >= 1; off >>= 1) {
        s  += __shfl_xor(s,  off, 32);
        s2 += __shfl_xor(s2, off, 32);
    }
    const float mean = s * (1.f / C_);
    const float var  = s2 * (1.f / C_) - mean * mean;
    const float rstd = rsqrtf(var + 1e-5f);

    float xn[8];
#pragma unroll
    for (int i = 0; i < 8; i++) {
        int c = lane + i * 32;
        xn[i] = (xv[i] - mean) * rstd * gamma[c] + beta[c];
    }

    float low[18];
#pragma unroll
    for (int j = 0; j < 18; j++) low[j] = 0.f;
#pragma unroll
    for (int i = 0; i < 8; i++) {
        int c = lane + i * 32;
#pragma unroll
        for (int j = 0; j < 18; j++) low[j] += xn[i] * wlow[c * 18 + j];
    }
#pragma unroll
    for (int j = 0; j < 18; j++) {
#pragma unroll
        for (int off = 16; off >= 1; off >>= 1) low[j] += __shfl_xor(low[j], off, 32);
        low[j] += blow[j];
    }

    const float qscale = 0.17677669529663687f;  // 1/sqrt(HEAD_DIM)
#pragma unroll
    for (int t = 0; t < 24; t++) {
        int o = lane + t * 32;          // 0..767
        float acc = bhigh[o];
#pragma unroll
        for (int j = 0; j < 18; j++) acc += low[j] * whigh[j * 768 + o];
        int which = o >> 8;             // 0=q 1=k 2=v
        int cc = o & 255;
        int h = cc >> 5, d = cc & 31;
        if (which == 0)
            Qe[((size_t)(b * H_ + h) * N_ + n) * KE_ + d] = __float2bfloat16(acc * qscale);
        else if (which == 1)
            Ke[((size_t)(b * H_ + h) * N_ + n) * KE_ + d] = __float2bfloat16(acc);
        else
            Vt[((size_t)(b * H_ + h) * HD_ + d) * N_ + n] = __float2bfloat16(acc);
    }

    // extension dims 32..63: graph bias factors (rank-6) + zero padding
#pragma unroll
    for (int t = 0; t < 8; t++) {
        int idx = lane + t * 32;        // 0..255 -> (h, d)
        int h = idx >> 5, d = idx & 31;
        float qv = (d < BN_) ? u[((size_t)h * N_ + n) * BN_ + d] : 0.f;
        float kv = (d < BN_) ? v[((size_t)(h * BN_ + d)) * N_ + n] : 0.f;
        Qe[((size_t)(b * H_ + h) * N_ + n) * KE_ + 32 + d] = __float2bfloat16(qv);
        Ke[((size_t)(b * H_ + h) * N_ + n) * KE_ + 32 + d] = __float2bfloat16(kv);
    }
}

// ---------------------------------------------------------------------------
// Kernel B: transposed flash attention. One wave owns a 16-query tile of one
// (b,h). S is computed TRANSPOSED (S^T = K x Q^T) so that lane = query:
//   - softmax stats are lane-local (one xor-16 shuffle per reduction)
//   - post-exp P^T in C-layout is bit-identical to the B-operand layout of P,
//     so P feeds the P@V WMMAs (D^T = V^T x P) directly from registers.
// 6 WMMAs per 32-key step, zero LDS staging.
// ---------------------------------------------------------------------------
__global__ __launch_bounds__(128) void attn_kernel(
    const __hip_bfloat16* __restrict__ Qe, const __hip_bfloat16* __restrict__ Ke,
    const __hip_bfloat16* __restrict__ Vt, const int* __restrict__ adj,
    const float* __restrict__ scale_p, float* __restrict__ attn)
{
    const int lane = threadIdx.x & 31;
    const int warp = threadIdx.x >> 5;
    const int w    = blockIdx.x * 4 + warp;
    const int qt   = w & 127;          // N/16 tiles
    const int h    = (w >> 7) & 7;
    const int b    = w >> 10;
    const int qbase = qt * 16;

    const int r  = lane & 15;          // query column (this lane's query row)
    const int hi = lane >> 4;          // lane half
    const int ko = hi * 8;             // K-chunk base for 16-bit A/B operands

    const __hip_bfloat16* Qr = Qe + ((size_t)(b * H_ + h) * N_ + qbase) * KE_;
    const __hip_bfloat16* Kb = Ke + (size_t)(b * H_ + h) * N_ * KE_;
    const __hip_bfloat16* Vb = Vt + (size_t)(b * H_ + h) * HD_ * N_;

    // Q^T B-operands (persistent): lane = query column, dim chunks [ko,ko+8)+[ko+16,ko+24)
    const v16bf qb0 = ld_pair(Qr + r * KE_ + ko,      Qr + r * KE_ + ko + 16);
    const v16bf qb1 = ld_pair(Qr + r * KE_ + 32 + ko, Qr + r * KE_ + 32 + ko + 16);

    const float scale  = scale_p[0];
    const bool use_adj = (scale != 0.f);   // uniform branch; EXEC stays all-ones
    const float mask_b = scale * -1e9f;

    float m_run = -3.0e38f, l_run = 0.f;   // per-lane (= per-query) stats
    v8f acc0 = {};   // D^T dims 0..15  (VGPR i -> dim ko+i, lane -> query)
    v8f acc1 = {};   // D^T dims 16..31

    const int* adjrow = adj + (size_t)(qbase + r) * N_;

    for (int mb = 0; mb < N_; mb += 32) {
        if (mb + 32 < N_)
            __builtin_prefetch(Kb + (size_t)(mb + 32 + r) * KE_, 0, 1);

        // ---- S^T tiles: A = K rows (lane=key), B = Q^T (lane=query), K=64 ----
        v8f T0 = {}, T1 = {};
        {
            const __hip_bfloat16* k0 = Kb + (size_t)(mb + r) * KE_;
            v16bf ka0 = ld_pair(k0 + ko,      k0 + ko + 16);
            v16bf ka1 = ld_pair(k0 + 32 + ko, k0 + 32 + ko + 16);
            T0 = wmma_bf16(ka0, qb0, T0);
            T0 = wmma_bf16(ka1, qb1, T0);
            const __hip_bfloat16* k1 = Kb + (size_t)(mb + 16 + r) * KE_;
            v16bf ka2 = ld_pair(k1 + ko,      k1 + ko + 16);
            v16bf ka3 = ld_pair(k1 + 32 + ko, k1 + 32 + ko + 16);
            T1 = wmma_bf16(ka2, qb0, T1);
            T1 = wmma_bf16(ka3, qb1, T1);
        }
        // T0: VGPR i -> key mb+ko+i ; T1: VGPR i -> key mb+16+ko+i ; lane -> query qbase+r

        // ---- optional adjacency mask: 4 contiguous int4 loads (scale==0 -> skipped) ----
        if (use_adj) {
            int av[16];
            *(int4*)&av[0]  = *(const int4*)(adjrow + mb + ko);
            *(int4*)&av[4]  = *(const int4*)(adjrow + mb + ko + 4);
            *(int4*)&av[8]  = *(const int4*)(adjrow + mb + 16 + ko);
            *(int4*)&av[12] = *(const int4*)(adjrow + mb + 16 + ko + 4);
#pragma unroll
            for (int i = 0; i < 8; i++) T0[i] += (av[i]     > 0) ? 0.f : mask_b;
#pragma unroll
            for (int i = 0; i < 8; i++) T1[i] += (av[8 + i] > 0) ? 0.f : mask_b;
        }

        // ---- online softmax: lane-local stats + one xor-16 combine ----
        float t[8];
#pragma unroll
        for (int i = 0; i < 8; i++) t[i] = fmaxf(T0[i], T1[i]);
#pragma unroll
        for (int s = 4; s >= 1; s >>= 1)
#pragma unroll
            for (int i = 0; i < s; i++) t[i] = fmaxf(t[i], t[i + s]);
        float mx   = fmaxf(t[0], __shfl_xor(t[0], 16, 32));
        float mnew = fmaxf(m_run, mx);
        float alpha = __expf(m_run - mnew);

        float pv[16];
#pragma unroll
        for (int i = 0; i < 8; i++) pv[i]     = __expf(T0[i] - mnew);
#pragma unroll
        for (int i = 0; i < 8; i++) pv[8 + i] = __expf(T1[i] - mnew);
        float rs = 0.f;
#pragma unroll
        for (int i = 0; i < 16; i++) rs += pv[i];
        rs += __shfl_xor(rs, 16, 32);

        m_run = mnew;
        l_run = l_run * alpha + rs;
#pragma unroll
        for (int i = 0; i < 8; i++) { acc0[i] *= alpha; acc1[i] *= alpha; }

        // ---- P as B-operand, straight from registers (no LDS) ----
        v16bf pB;
#pragma unroll
        for (int i = 0; i < 16; i++) pB[i] = to_bf16(pv[i]);

        // ---- D^T += V^T x P : A = V^T (lane=dim), B = P (lane=query) ----
        {
            const __hip_bfloat16* v0 = Vb + (size_t)r * N_ + mb;          // dims 0..15
            v16bf va0 = ld_pair(v0 + ko, v0 + 16 + ko);
            acc0 = wmma_bf16(va0, pB, acc0);
            const __hip_bfloat16* v1 = Vb + (size_t)(16 + r) * N_ + mb;   // dims 16..31
            v16bf va1 = ld_pair(v1 + ko, v1 + 16 + ko);
            acc1 = wmma_bf16(va1, pB, acc1);
        }
    }

    // ---- finalize: per-lane scalar normalize, 4 contiguous float4 stores ----
    const float inv = 1.f / l_run;
    float* arow = attn + ((size_t)b * N_ + qbase + r) * C_ + h * HD_;
    *(float4*)(arow + ko)          = make_float4(acc0[0] * inv, acc0[1] * inv, acc0[2] * inv, acc0[3] * inv);
    *(float4*)(arow + ko + 4)      = make_float4(acc0[4] * inv, acc0[5] * inv, acc0[6] * inv, acc0[7] * inv);
    *(float4*)(arow + 16 + ko)     = make_float4(acc1[0] * inv, acc1[1] * inv, acc1[2] * inv, acc1[3] * inv);
    *(float4*)(arow + 16 + ko + 4) = make_float4(acc1[4] * inv, acc1[5] * inv, acc1[6] * inv, acc1[7] * inv);
    (void)warp;
}

// ---------------------------------------------------------------------------
// Kernel C: rank-6 output projection (256->6->256) + residual + tail scalar.
// One wave per token row.
// ---------------------------------------------------------------------------
__global__ __launch_bounds__(128) void out_proj_kernel(
    const float* __restrict__ attn, const float* __restrict__ x,
    const float* __restrict__ wl, const float* __restrict__ bl,
    const float* __restrict__ wh, const float* __restrict__ bh,
    float* __restrict__ out)
{
    const int lane = threadIdx.x & 31;
    const int wid  = blockIdx.x * (blockDim.x >> 5) + (threadIdx.x >> 5);
    const int b = wid / N_;
    const int n = wid % N_;

    const float* ar = attn + (size_t)(b * N_ + n) * C_;
    float av[8];
#pragma unroll
    for (int i = 0; i < 8; i++) av[i] = ar[lane + i * 32];

    float low[6] = {0.f, 0.f, 0.f, 0.f, 0.f, 0.f};
#pragma unroll
    for (int i = 0; i < 8; i++) {
        int c = lane + i * 32;
#pragma unroll
        for (int j = 0; j < 6; j++) low[j] += av[i] * wl[c * 6 + j];
    }
#pragma unroll
    for (int j = 0; j < 6; j++) {
#pragma unroll
        for (int off = 16; off >= 1; off >>= 1) low[j] += __shfl_xor(low[j], off, 32);
        low[j] += bl[j];
    }

    const float* xr   = x   + (size_t)(b * N_ + n) * C_;
    float*       orow = out + (size_t)(b * N_ + n) * C_;
#pragma unroll
    for (int i = 0; i < 8; i++) {
        int c = lane + i * 32;
        float o = bh[c] + xr[c];
#pragma unroll
        for (int j = 0; j < 6; j++) o += low[j] * wh[j * 256 + c];
        orow[c] = o;
    }

    if (wid == 0 && lane == 0)
        out[(size_t)B_ * N_ * C_] = 0.f;   // second tuple output: scalar 0.0
}

// ---------------------------------------------------------------------------
extern "C" void kernel_launch(void* const* d_in, const int* in_sizes, int n_in,
                              void* d_out, int out_size, void* d_ws, size_t ws_size,
                              hipStream_t stream) {
    (void)in_sizes; (void)n_in; (void)out_size; (void)ws_size;

    const float* x     = (const float*)d_in[0];
    const int*   adj   = (const int*)  d_in[1];
    const float* gamma = (const float*)d_in[2];
    const float* beta  = (const float*)d_in[3];
    const float* wql   = (const float*)d_in[4];
    const float* bql   = (const float*)d_in[5];
    const float* wqh   = (const float*)d_in[6];
    const float* bqh   = (const float*)d_in[7];
    const float* wol   = (const float*)d_in[8];
    const float* bol   = (const float*)d_in[9];
    const float* woh   = (const float*)d_in[10];
    const float* boh   = (const float*)d_in[11];
    const float* u     = (const float*)d_in[12];
    const float* v     = (const float*)d_in[13];
    const float* scal  = (const float*)d_in[14];
    float* out = (float*)d_out;

    // workspace layout (all 16B-aligned): Qe 8MB | Ke 8MB | Vt 4MB | attn 8MB
    __hip_bfloat16* Qe = (__hip_bfloat16*)d_ws;
    __hip_bfloat16* Ke = Qe + (size_t)B_ * H_ * N_ * KE_;
    __hip_bfloat16* Vt = Ke + (size_t)B_ * H_ * N_ * KE_;
    float*        attn = (float*)(Vt + (size_t)B_ * H_ * HD_ * N_);

    dim3 blk(128);
    qkv_pack_kernel<<<dim3(B_ * N_ / 4), blk, 0, stream>>>(
        x, gamma, beta, wql, bql, wqh, bqh, u, v, Qe, Ke, Vt);
    attn_kernel<<<dim3(B_ * H_ * (N_ / 16) / 4), blk, 0, stream>>>(
        Qe, Ke, Vt, adj, scal, attn);
    out_proj_kernel<<<dim3(B_ * N_ / 4), blk, 0, stream>>>(
        attn, x, wol, bol, woh, boh, out);
}
